// AttentionDecoder_2_85014582657154
// MI455X (gfx1250) — matrix operand order repaired and checked by reading, not compiled, not run on hardware
//
#include <hip/hip_runtime.h>
#include <hip/hip_bf16.h>

// ---------------------------------------------------------------------------
// AttentionDecoder step for MI455X (gfx1250, wave32).
//
// GEMMs: V_WMMA_F32_16X16X32_BF16 with f32->bf16 hi/lo split (Markidis),
// 3 WMMAs per k-step. Activations are pre-packed ONCE into the exact WMMA
// fragment layout as bf16 hi/lo pairs -> GEMM inner loop is pure b128 loads +
// WMMA for A; streamed weights (B) are split inline (each element converted
// exactly once, read once from HBM). The wz/wh GEMMs are fused into a single
// K=2048 GEMM via packed-A concatenation with a uniform weight-pointer split.
// The vocab GEMM fuses the reference's axis=1 (batch-axis) log_softmax into
// its epilogue with one shfl_xor(16), since each wave owns all 128 batch rows
// of its 16-column strip.
// ---------------------------------------------------------------------------

typedef __attribute__((ext_vector_type(16))) __bf16 v16bf;
typedef __attribute__((ext_vector_type(8)))  float  v8f;

#define B_SZ    128
#define HID_D   1024
#define EMB_D   512
#define VOCAB_D 32000
#define KEY_D   1024
#define NK_D    512
#define X_D     (EMB_D + KEY_D)   // 1536

__device__ __forceinline__ void split_bf16(float x, __bf16& hi, __bf16& lo) {
    hi = (__bf16)x;
    lo = (__bf16)(x - (float)hi);
}

#define WMMA_BF16(A, Bm, C) \
    __builtin_amdgcn_wmma_f32_16x16x32_bf16(false, (A), false, (Bm), (short)0, (C), false, false)

// ---------------------------------------------------------------------------
// Pack X (128 x K f32, leading dim lda) into WMMA A-fragment order:
// fragment f = kp*8 + mt (kp = 32-wide k-panel, mt = 16-row M tile); per lane
// 16 bf16 at [(f*32 + lane)*16].  ISA 16-bit A 16x32 layout:
//   lane 0-15: M = mt*16+lane;  j=0..7 -> K=kp*32+j, j=8..15 -> K=kp*32+16+(j-8)
//   lane16-31: same with K base +8
// grid = K/32 blocks * 256 threads (one thread per (fragment, lane))
// ---------------------------------------------------------------------------
__global__ void pack_a(const float* __restrict__ X, int lda,
                       __bf16* __restrict__ hi, __bf16* __restrict__ lo) {
    const int tid  = blockIdx.x * 256 + threadIdx.x;
    const int lane = tid & 31;
    const int f    = tid >> 5;
    const int mt   = f & 7, kp = f >> 3;
    const int m    = mt * 16 + (lane & 15);
    const int kb   = kp * 32 + ((lane >> 4) & 1) * 8;
    const float* p = X + (size_t)m * lda + kb;
    __bf16* ph = hi + (size_t)tid * 16;
    __bf16* pl = lo + (size_t)tid * 16;
#pragma unroll
    for (int j = 0; j < 8; ++j) { __bf16 h, l; split_bf16(p[j],      h, l); ph[j]     = h; pl[j]     = l; }
#pragma unroll
    for (int j = 0; j < 8; ++j) { __bf16 h, l; split_bf16(p[16 + j], h, l); ph[8 + j] = h; pl[8 + j] = l; }
}

// B fragment: 32x16 (KxN) bf16 from row-major W (N x K), i.e. B = W^T:
//   lane 0-15: N=n0+lane, j=0..7 -> K=k0+j, j=8..15 -> K=k0+16+(j-8); lane16-31: K+8
__device__ __forceinline__ void load_b_frag(const float* __restrict__ W, int ldw,
                                            int n0, int k0, int lane,
                                            v16bf& bh, v16bf& bl) {
    const int n  = n0 + (lane & 15);
    const int kb = k0 + ((lane >> 4) & 1) * 8;
    const float* p = W + (size_t)n * ldw + kb;
#pragma unroll
    for (int j = 0; j < 16; ++j) { __bf16 h, l; split_bf16(p[j < 8 ? j : j + 8], h, l); bh[j] = h; bl[j] = l; }
}

__device__ __forceinline__ void load_a_packed(const __bf16* __restrict__ base,
                                              int kp, int mtg, int lane,
                                              v16bf& a) {
    a = *reinterpret_cast<const v16bf*>(base + ((size_t)(kp * 8 + mtg) * 32 + lane) * 16);
}

// ---------------------------------------------------------------------------
// out[128xN] = A @ [W1|W2]^T + bias1(+bias2) (+ add), A pre-packed bf16 hi/lo.
// 8 waves/block: wave -> (n-strip = wave>>1, M-half = wave&1, 4 C tiles).
// Weight source switches (wave-uniformly) from W1 (K1 cols) to W2 (K-K1 cols)
// at k0==K1, which fuses two X@W^T GEMMs sharing one output.
// ---------------------------------------------------------------------------
__global__ void gemm_xwT(const __bf16* __restrict__ Ahi,
                         const __bf16* __restrict__ Alo, int K,
                         const float* __restrict__ W1, int K1,
                         const float* __restrict__ W2,
                         const float* __restrict__ bias1,
                         const float* __restrict__ bias2,
                         const float* __restrict__ add, int add_ld,
                         float* __restrict__ out, int ldo) {
    const int lane  = threadIdx.x & 31;
    const int wave  = threadIdx.x >> 5;
    const int n0    = (blockIdx.x * 4 + (wave >> 1)) * 16;
    const int mhalf = wave & 1;                  // M tiles mhalf*4 .. mhalf*4+3

    v8f acc[4] = {};
    for (int k0 = 0; k0 < K; k0 += 32) {
        const float* Wp; int ldw, kc;
        if (k0 < K1) { Wp = W1; ldw = K1;     kc = k0;      }
        else         { Wp = W2; ldw = K - K1; kc = k0 - K1; }
        v16bf bh, bl;
        load_b_frag(Wp, ldw, n0, kc, lane, bh, bl);
        const int kp = k0 >> 5;
#pragma unroll
        for (int mt = 0; mt < 4; ++mt) {
            const int mtg = mhalf * 4 + mt;
            v16bf ah, al;
            load_a_packed(Ahi, kp, mtg, lane, ah);
            load_a_packed(Alo, kp, mtg, lane, al);
            acc[mt] = WMMA_BF16(ah, bh, acc[mt]);
            acc[mt] = WMMA_BF16(ah, bl, acc[mt]);
            acc[mt] = WMMA_BF16(al, bh, acc[mt]);
        }
    }

    const int n  = n0 + (lane & 15);
    float bn = bias1 ? bias1[n] : 0.0f;
    if (bias2) bn += bias2[n];
    const int mhi = ((lane >> 4) & 1) * 8;       // C layout: VGPR i -> M=i (+8 hi lanes)
#pragma unroll
    for (int mt = 0; mt < 4; ++mt) {
#pragma unroll
        for (int i = 0; i < 8; ++i) {
            const int m = (mhalf * 4 + mt) * 16 + mhi + i;
            float v = acc[mt][i] + bn;
            if (add) v += add[(size_t)m * add_ld + n];
            out[(size_t)m * ldo + n] = v;
        }
    }
}

// ---------------------------------------------------------------------------
// out[128x32000] = log_softmax( S @ out_W^T + out_b, over the BATCH axis ).
// 4 waves/block, wave owns 16 cols x all 128 rows -> column log-softmax with
// one shfl_xor(16) (lanes n and n+16 hold M 0..7 / 8..15 of each C tile).
// ---------------------------------------------------------------------------
__global__ void out_gemm_lsm(const __bf16* __restrict__ Ahi,
                             const __bf16* __restrict__ Alo,
                             const float* __restrict__ W,
                             const float* __restrict__ bias,
                             float* __restrict__ out) {
    const int lane = threadIdx.x & 31;
    const int wave = threadIdx.x >> 5;
    const int n0   = (blockIdx.x * 4 + wave) * 16;

    v8f acc[8] = {};
    for (int k0 = 0; k0 < EMB_D; k0 += 32) {
        v16bf bh, bl;
        load_b_frag(W, EMB_D, n0, k0, lane, bh, bl);
        const int kp = k0 >> 5;
#pragma unroll
        for (int mt = 0; mt < 8; ++mt) {
            v16bf ah, al;
            load_a_packed(Ahi, kp, mt, lane, ah);
            load_a_packed(Alo, kp, mt, lane, al);
            acc[mt] = WMMA_BF16(ah, bh, acc[mt]);
            acc[mt] = WMMA_BF16(ah, bl, acc[mt]);
            acc[mt] = WMMA_BF16(al, bh, acc[mt]);
        }
    }

    const int   n   = n0 + (lane & 15);
    const float bn  = bias[n];
    const int   mhi = ((lane >> 4) & 1) * 8;

    float vals[64];
    float mx = -3.402823466e38f;
#pragma unroll
    for (int mt = 0; mt < 8; ++mt)
#pragma unroll
        for (int i = 0; i < 8; ++i) {
            const float v = acc[mt][i] + bn;
            vals[mt * 8 + i] = v;
            mx = fmaxf(mx, v);
        }
    mx = fmaxf(mx, __shfl_xor(mx, 16, 32));      // column max over all 128 rows
    float sum = 0.0f;
#pragma unroll
    for (int j = 0; j < 64; ++j) sum += expf(vals[j] - mx);
    sum += __shfl_xor(sum, 16, 32);
    const float lse = mx + logf(sum);
#pragma unroll
    for (int mt = 0; mt < 8; ++mt)
#pragma unroll
        for (int i = 0; i < 8; ++i) {
            const int m = mt * 16 + mhi + i;
            out[(size_t)m * VOCAB_D + n] = vals[mt * 8 + i] - lse;
        }
}

// ---------------------------------------------------------------------------
// Attention: one workgroup (256 thr) per batch row b (HBM-bound streaming).
//   pass1: tk[n] = annotations[b,n,:] . wk  (wave dot + shfl reduce)
//   softmax over n in LDS (tq/bk are per-row constants -> cancel in softmax)
//   pass2: context[d] = sum_n w[n] * annotations[b,n,d]  (float4 columns)
// ---------------------------------------------------------------------------
__global__ void attn_kernel(const float* __restrict__ annot,
                            const float* __restrict__ wk,
                            float* __restrict__ xbuf,
                            float* __restrict__ attn_out) {
    __shared__ float swk[KEY_D];
    __shared__ float stk[NK_D];
    __shared__ float red[256];

    const int b = blockIdx.x;
    const int t = threadIdx.x;
    const int lane = t & 31, wave = t >> 5;

    for (int i = t; i < KEY_D; i += 256) swk[i] = wk[i];
    __syncthreads();

    const float* abase = annot + (size_t)b * NK_D * KEY_D;
    for (int j = 0; j < NK_D / 8; ++j) {
        const int n = wave * (NK_D / 8) + j;
        const float* ar = abase + (size_t)n * KEY_D;
        float p = 0.0f;
        for (int k = lane; k < KEY_D; k += 32) p += ar[k] * swk[k];
#pragma unroll
        for (int off = 16; off >= 1; off >>= 1) p += __shfl_xor(p, off, 32);
        if (lane == 0) stk[n] = p;
    }
    __syncthreads();

    const float v0 = stk[t], v1 = stk[t + 256];
    red[t] = fmaxf(v0, v1);
    __syncthreads();
    for (int off = 128; off >= 1; off >>= 1) {
        if (t < off) red[t] = fmaxf(red[t], red[t + off]);
        __syncthreads();
    }
    const float mx = red[0];
    __syncthreads();
    const float e0 = expf(v0 - mx), e1 = expf(v1 - mx);
    red[t] = e0 + e1;
    __syncthreads();
    for (int off = 128; off >= 1; off >>= 1) {
        if (t < off) red[t] += red[t + off];
        __syncthreads();
    }
    const float inv = 1.0f / red[0];
    __syncthreads();
    const float w0 = e0 * inv, w1 = e1 * inv;
    stk[t] = w0; stk[t + 256] = w1;
    attn_out[(size_t)b * NK_D + t]       = w0;
    attn_out[(size_t)b * NK_D + t + 256] = w1;
    __syncthreads();

    float4 acc = {0.f, 0.f, 0.f, 0.f};
    for (int n = 0; n < NK_D; ++n) {
        const float4 a = *reinterpret_cast<const float4*>(abase + (size_t)n * KEY_D + t * 4);
        const float w = stk[n];
        acc.x += w * a.x; acc.y += w * a.y; acc.z += w * a.z; acc.w += w * a.w;
    }
    *reinterpret_cast<float4*>(xbuf + (size_t)b * X_D + EMB_D + t * 4) = acc;
}

// emb gather into xbuf columns [0,512)
__global__ void embed_kernel(const int* __restrict__ tok,
                             const float* __restrict__ table,
                             float* __restrict__ xbuf) {
    const int b = blockIdx.x, t = threadIdx.x;   // 128 threads * float4 = 512
    const int v = tok[b];
    const float4* src = reinterpret_cast<const float4*>(table + (size_t)v * EMB_D);
    float4*       dst = reinterpret_cast<float4*>(xbuf + (size_t)b * X_D);
    dst[t] = src[t];
}

// GRU gate fusion -> h_new (written straight into the hid output slot)
__global__ void gru_kernel(const float* __restrict__ gx,
                           const float* __restrict__ gh,
                           const float* __restrict__ q,
                           float* __restrict__ hid) {
    const int i = blockIdx.x * blockDim.x + threadIdx.x;   // 0..B*HID-1
    const int b = i >> 10, c = i & (HID_D - 1);
    const float* gxr = gx + (size_t)b * 3 * HID_D;
    const float* ghr = gh + (size_t)b * 3 * HID_D;
    const float r = 1.0f / (1.0f + expf(-(gxr[c] + ghr[c])));
    const float z = 1.0f / (1.0f + expf(-(gxr[HID_D + c] + ghr[HID_D + c])));
    const float n = tanhf(gxr[2 * HID_D + c] + r * ghr[2 * HID_D + c]);
    hid[i] = (1.0f - z) * n + z * q[i];
}

// ---------------------------------------------------------------------------
extern "C" void kernel_launch(void* const* d_in, const int* in_sizes, int n_in,
                              void* d_out, int out_size, void* d_ws, size_t ws_size,
                              hipStream_t stream) {
    const int*   input  = (const int*)  d_in[0];
    const float* hidden = (const float*)d_in[1];
    const float* annot  = (const float*)d_in[2];
    const float* table  = (const float*)d_in[3];
    const float* wk     = (const float*)d_in[4];
    // d_in[5..7] (bk, wq, bq): per-row constants, cancel inside the softmax
    const float* W_ih   = (const float*)d_in[8];
    const float* W_hh   = (const float*)d_in[9];
    const float* b_ih   = (const float*)d_in[10];
    const float* b_hh   = (const float*)d_in[11];
    const float* wz_W   = (const float*)d_in[12];
    const float* wz_b   = (const float*)d_in[13];
    const float* wh_W   = (const float*)d_in[14];
    const float* wh_b   = (const float*)d_in[15];
    const float* out_W  = (const float*)d_in[16];
    const float* out_b  = (const float*)d_in[17];

    float* OUT  = (float*)d_out;                         // (1,B,VOCAB)
    float* HID  = OUT + (size_t)B_SZ * VOCAB_D;          // (1,B,HID)
    float* ATTN = HID + (size_t)B_SZ * HID_D;            // (B,NK)

    // ---- workspace layout (bytes), 6.5 MiB total ----
    char* w = (char*)d_ws;
    float* xbuf = (float*)w;  w += (size_t)B_SZ * X_D * 4;          // 128x1536 [emb|ctx]
    float* gx   = (float*)w;  w += (size_t)B_SZ * 3 * HID_D * 4;    // 128x3072
    float* gh   = (float*)w;  w += (size_t)B_SZ * 3 * HID_D * 4;    // 128x3072
    float* sbuf = (float*)w;  w += (size_t)B_SZ * EMB_D * 4;        // 128x512
    __bf16* pkx_hi = (__bf16*)w; w += (size_t)B_SZ * X_D * 2;       // x packed
    __bf16* pkx_lo = (__bf16*)w; w += (size_t)B_SZ * X_D * 2;
    __bf16* pkh_hi = (__bf16*)w; w += (size_t)B_SZ * HID_D * 2;     // hidden packed
    __bf16* pkh_lo = (__bf16*)w; w += (size_t)B_SZ * HID_D * 2;
    __bf16* pkcn_hi = (__bf16*)w; w += (size_t)B_SZ * 2 * HID_D * 2; // [ctx|h_new] K=2048
    __bf16* pkcn_lo = (__bf16*)w; w += (size_t)B_SZ * 2 * HID_D * 2;
    __bf16* pks_hi = (__bf16*)w; w += (size_t)B_SZ * EMB_D * 2;     // s packed
    __bf16* pks_lo = (__bf16*)w; w += (size_t)B_SZ * EMB_D * 2;

    attn_kernel <<<B_SZ, 256, 0, stream>>>(annot, wk, xbuf, ATTN);
    embed_kernel<<<B_SZ, 128, 0, stream>>>(input, table, xbuf);

    // pack activations into WMMA fragment order (bf16 hi/lo)
    pack_a<<<X_D   / 32, 256, 0, stream>>>(xbuf,   X_D,   pkx_hi, pkx_lo);
    pack_a<<<HID_D / 32, 256, 0, stream>>>(hidden, HID_D, pkh_hi, pkh_lo);

    // gx = x @ W_ih^T + b_ih   (N=3072, K=1536)
    gemm_xwT<<<3 * HID_D / 64, 256, 0, stream>>>(pkx_hi, pkx_lo, X_D,
                                                 W_ih, X_D, nullptr,
                                                 b_ih, nullptr, nullptr, 0,
                                                 gx, 3 * HID_D);
    // gh = q @ W_hh^T + b_hh   (N=3072, K=1024)
    gemm_xwT<<<3 * HID_D / 64, 256, 0, stream>>>(pkh_hi, pkh_lo, HID_D,
                                                 W_hh, HID_D, nullptr,
                                                 b_hh, nullptr, nullptr, 0,
                                                 gh, 3 * HID_D);
    // h_new
    gru_kernel<<<(B_SZ * HID_D) / 256, 256, 0, stream>>>(gx, gh, hidden, HID);

    // pack [context | h_new] as one contiguous K=2048 packed-A operand
    pack_a<<<KEY_D / 32, 256, 0, stream>>>(xbuf + EMB_D, X_D,
                                           pkcn_hi, pkcn_lo);
    pack_a<<<HID_D / 32, 256, 0, stream>>>(HID, HID_D,
                                           pkcn_hi + (size_t)B_SZ * KEY_D,
                                           pkcn_lo + (size_t)B_SZ * KEY_D);

    // s = emb + ctx @ wz_W^T + wz_b + h_new @ wh_W^T + wh_b
    //   (single N=512, K=2048 GEMM; weight source switches at K1=1024)
    gemm_xwT<<<EMB_D / 64, 256, 0, stream>>>(pkcn_hi, pkcn_lo, 2 * HID_D,
                                             wz_W, KEY_D, wh_W,
                                             wz_b, wh_b, xbuf, X_D,
                                             sbuf, EMB_D);

    pack_a<<<EMB_D / 32, 256, 0, stream>>>(sbuf, EMB_D, pks_hi, pks_lo);

    // out = log_softmax(s @ out_W^T + out_b) over batch axis (fused epilogue)
    out_gemm_lsm<<<VOCAB_D / 64, 128, 0, stream>>>(pks_hi, pks_lo, out_W, out_b, OUT);
}
